// TrajectoryDecoder_18442589570231
// MI455X (gfx1250) — matrix operand order repaired and checked by reading, not compiled, not run on hardware
//
#include <hip/hip_runtime.h>

// ---- problem constants ----
#define LL   6
#define NN   100
#define DD   512
#define NHH  8
#define PP   12
#define KK   4
#define TT   4
#define CC   64
#define HFF  64
#define WFF  160
#define BB   4
#define FEATD (KK*TT*CC*8)   // 8192
#define ROWS  (BB*NN)        // 400
#define DHH   64             // head dim

typedef __attribute__((ext_vector_type(16))) _Float16 v16h;
typedef __attribute__((ext_vector_type(8)))  _Float16 v8h;
typedef __attribute__((ext_vector_type(8)))  float    v8f;

// ---------------------------------------------------------------------------
// GEMM: C(M x Nc, f32) = A(M x K, f16 row-major) * Bt^T + bias (+resid)(+relu)
// Bt stored transposed: (Npad x K) f16, row n = column n of W, zero-padded.
// One wave computes a 16x64 tile: 4 accumulators share one A fragment per
// k-step -> 4 v_wmma_f32_16x16x32_f16 per A load, 4x less A traffic.
// M % 16 == 0, K % 32 == 0, Npad % 64 == 0; only N stores are guarded.
// ---------------------------------------------------------------------------
__global__ __launch_bounds__(32) void td_gemm16(
    const _Float16* __restrict__ A, const _Float16* __restrict__ Bt,
    const float* __restrict__ bias, const float* __restrict__ resid,
    float* __restrict__ C, _Float16* __restrict__ C16,
    int M, int K, int Nc, int relu)
{
  int lane = threadIdx.x;
  int tn = blockIdx.x * 64, tm = blockIdx.y * 16;
  int l15 = lane & 15, lhi = lane >> 4;

  // A: lanes 0-15 carry K 0-15, lanes 16-31 carry K 16-31 (contiguous 16)
  const _Float16* Ap = A  + (size_t)(tm + l15) * K + lhi * 16;
  // B: lanes 0-15 carry K 0-15, lanes 16-31 carry K 16-31 (contiguous 16)
  const _Float16* Bp = Bt + (size_t)(tn + l15) * K + lhi * 16;
  const size_t bstr = (size_t)16 * K;          // 16 N-rows of Bt

  v8f acc[4] = {{}, {}, {}, {}};
  for (int k0 = 0; k0 < K; k0 += 32) {
    v8h a0 = *(const v8h*)(Ap + k0);
    v8h a1 = *(const v8h*)(Ap + k0 + 8);
    v16h av = __builtin_shufflevector(a0, a1, 0,1,2,3,4,5,6,7,8,9,10,11,12,13,14,15);
    v16h b0 = *(const v16h*)(Bp + k0);
    v16h b1 = *(const v16h*)(Bp + bstr + k0);
    v16h b2 = *(const v16h*)(Bp + 2 * bstr + k0);
    v16h b3 = *(const v16h*)(Bp + 3 * bstr + k0);
    acc[0] = __builtin_amdgcn_wmma_f32_16x16x32_f16(false, av, false, b0, (short)0, acc[0], false, false);
    acc[1] = __builtin_amdgcn_wmma_f32_16x16x32_f16(false, av, false, b1, (short)0, acc[1], false, false);
    acc[2] = __builtin_amdgcn_wmma_f32_16x16x32_f16(false, av, false, b2, (short)0, acc[2], false, false);
    acc[3] = __builtin_amdgcn_wmma_f32_16x16x32_f16(false, av, false, b3, (short)0, acc[3], false, false);
  }

#pragma unroll
  for (int s = 0; s < 4; ++s) {
    int n = tn + 16 * s + l15;
    if (n < Nc) {
      float bvl = bias ? bias[n] : 0.0f;
#pragma unroll
      for (int r = 0; r < 8; ++r) {
        int m = tm + lhi * 8 + r;             // C layout: vgpr r -> row r / r+8
        float v = acc[s][r] + bvl;
        if (resid) v += resid[(size_t)m * Nc + n];
        if (relu)  v = fmaxf(v, 0.0f);
        C[(size_t)m * Nc + n] = v;
        if (C16) C16[(size_t)m * Nc + n] = (_Float16)v;
      }
    }
  }
}

// ---------------------------------------------------------------------------
// Weight convert + transpose + pad: W (K x Nc, f32) -> Wt (Npad x K, f16)
// ---------------------------------------------------------------------------
__global__ __launch_bounds__(256) void td_cvtT(
    const float* __restrict__ W, _Float16* __restrict__ Wt,
    int K, int Nc, int Npad)
{
  int i = blockIdx.x * 256 + threadIdx.x;
  if (i >= Npad * K) return;
  int nrow = i / K, k = i - nrow * K;
  Wt[(size_t)nrow * K + k] =
      (nrow < Nc) ? (_Float16)W[(size_t)k * Nc + nrow] : (_Float16)0.0f;
}

// ---------------------------------------------------------------------------
// q init: broadcast queries0 (N x D) to (B x N x D), f32 + f16
// ---------------------------------------------------------------------------
__global__ __launch_bounds__(256) void td_initq(
    const float* __restrict__ q0, float* __restrict__ q32,
    _Float16* __restrict__ q16)
{
  int i = blockIdx.x * 256 + threadIdx.x;
  if (i >= BB * NN * DD) return;
  float v = q0[i % (NN * DD)];
  q32[i] = v;
  q16[i] = (_Float16)v;
}

// ---------------------------------------------------------------------------
// LayerNorm over D=512, one wave per row; writes f32 and f16 copies.
// ---------------------------------------------------------------------------
__global__ __launch_bounds__(256) void td_ln(
    const float* __restrict__ X, const float* __restrict__ g,
    const float* __restrict__ bta, float* __restrict__ Y,
    _Float16* __restrict__ Y16, int rows)
{
  int wid = threadIdx.x >> 5, lane = threadIdx.x & 31;
  int row = blockIdx.x * 8 + wid;
  if (row >= rows) return;
  const float* xp = X + (size_t)row * DD;
  float vals[16];
  float s = 0.0f;
#pragma unroll
  for (int j = 0; j < 16; ++j) { vals[j] = xp[lane + 32 * j]; s += vals[j]; }
#pragma unroll
  for (int o = 16; o >= 1; o >>= 1) s += __shfl_xor(s, o, 32);
  float mean = s * (1.0f / DD);
  float vs = 0.0f;
#pragma unroll
  for (int j = 0; j < 16; ++j) { float d = vals[j] - mean; vs += d * d; }
#pragma unroll
  for (int o = 16; o >= 1; o >>= 1) vs += __shfl_xor(vs, o, 32);
  float rstd = rsqrtf(vs * (1.0f / DD) + 1e-5f);
#pragma unroll
  for (int j = 0; j < 16; ++j) {
    int d = lane + 32 * j;
    float yv = (vals[j] - mean) * rstd * g[d] + bta[d];
    Y[(size_t)row * DD + d] = yv;
    if (Y16) Y16[(size_t)row * DD + d] = (_Float16)yv;
  }
}

// ---------------------------------------------------------------------------
// Attention core: one wave per (b, h, query). scores -> softmax -> P@V.
// Q rows: Qb + (b*sq+qi)*ldq + qoff + h*64 ;  K/V rows in KVb with koff/voff.
// Output O (f16), row (b*sq+qi)*ldo + h*64.
// ---------------------------------------------------------------------------
__global__ __launch_bounds__(256) void td_attn(
    const float* __restrict__ Qb, int ldq, int qoff,
    const float* __restrict__ KVb, int ldkv, int koff, int voff,
    _Float16* __restrict__ O, int ldo,
    int sq, int skv, float scale)
{
  __shared__ float probs[8][128];
  int wid = threadIdx.x >> 5, lane = threadIdx.x & 31;
  int gw = blockIdx.x * 8 + wid;             // grid sized exactly: B*NH*sq/8
  int qi = gw % sq;
  int h  = (gw / sq) % NHH;
  int b  = gw / (sq * NHH);

  const float4* qp = (const float4*)(Qb + (size_t)(b * sq + qi) * ldq + qoff + h * DHH);
  float4 qr[16];
#pragma unroll
  for (int j = 0; j < 16; ++j) qr[j] = qp[j];

  float sc[4];
#pragma unroll
  for (int t = 0; t < 4; ++t) {
    int j = lane + 32 * t;
    float s = -1e30f;
    if (j < skv) {
      const float4* kp = (const float4*)(KVb + (size_t)(b * skv + j) * ldkv + koff + h * DHH);
      float acc = 0.0f;
#pragma unroll
      for (int d = 0; d < 16; ++d) {
        float4 k4 = kp[d];
        acc += qr[d].x * k4.x + qr[d].y * k4.y + qr[d].z * k4.z + qr[d].w * k4.w;
      }
      s = acc * scale;
    }
    sc[t] = s;
  }
  float m = fmaxf(fmaxf(sc[0], sc[1]), fmaxf(sc[2], sc[3]));
#pragma unroll
  for (int o = 16; o >= 1; o >>= 1) m = fmaxf(m, __shfl_xor(m, o, 32));
  float es[4], ssum = 0.0f;
#pragma unroll
  for (int t = 0; t < 4; ++t) {
    int j = lane + 32 * t;
    es[t] = (j < skv) ? __expf(sc[t] - m) : 0.0f;
    ssum += es[t];
  }
#pragma unroll
  for (int o = 16; o >= 1; o >>= 1) ssum += __shfl_xor(ssum, o, 32);
  float inv = 1.0f / ssum;
#pragma unroll
  for (int t = 0; t < 4; ++t) probs[wid][lane + 32 * t] = es[t] * inv;

  int d0 = lane * 2;
  float o0 = 0.0f, o1 = 0.0f;
  for (int j = 0; j < skv; ++j) {
    float pw = probs[wid][j];
    float2 v2 = *(const float2*)(KVb + (size_t)(b * skv + j) * ldkv + voff + h * DHH + d0);
    o0 += pw * v2.x;
    o1 += pw * v2.y;
  }
  _Float16* op = O + (size_t)(b * sq + qi) * ldo + h * DHH + d0;
  op[0] = (_Float16)o0;
  op[1] = (_Float16)o1;
}

// ---------------------------------------------------------------------------
// Box-corner projection + bilinear sampling.
// One thread per (b, n, k, corner); loops the 256 (t,c) channels.
// Output feats16 (B*N, 8192) f16, index k*2048 + tc*8 + corner.
// ---------------------------------------------------------------------------
__global__ __launch_bounds__(256) void td_sample(
    const float* __restrict__ traj, const float* __restrict__ feat,
    const float* __restrict__ intr, const float* __restrict__ extr,
    _Float16* __restrict__ outf)
{
  int idx = blockIdx.x * 256 + threadIdx.x;     // < B*N*K*8 = 12800
  if (idx >= BB * NN * KK * 8) return;
  int corner = idx & 7;
  int k = (idx >> 3) & 3;
  int rem = idx >> 5;
  int n = rem % NN, b = rem / NN;

  const float* t = traj + (size_t)(b * NN + n) * PP;
  float cx = t[0], cy = t[1], cz = t[2];
  float yaw = t[7];
  float dx = t[8], dy = t[9], dz = t[10];
  float sgx = (corner & 4) ? -1.0f : 1.0f;
  float sgy = (corner & 2) ? -1.0f : 1.0f;
  float sgz = (corner & 1) ? -1.0f : 1.0f;
  float ox = 0.5f * dx * sgx, oy = 0.5f * dy * sgy, oz = 0.5f * dz * sgz;
  float cyw = cosf(yaw), syw = sinf(yaw);
  float px = cx + cyw * ox - syw * oy;
  float py = cy + syw * ox + cyw * oy;
  float pz = cz + oz;

  const float* E = extr + k * 16;
  const float* I = intr + k * 9;
  float camx = E[0] * px + E[1] * py + E[2]  * pz + E[3];
  float camy = E[4] * px + E[5] * py + E[6]  * pz + E[7];
  float camz = E[8] * px + E[9] * py + E[10] * pz + E[11];
  float ix = I[0] * camx + I[1] * camy + I[2] * camz;
  float iy = I[3] * camx + I[4] * camy + I[5] * camz;
  float iz = I[6] * camx + I[7] * camy + I[8] * camz;
  float z = iz;
  if (fabsf(z) < 1e-3f) z = (z >= 0.0f) ? 1e-3f : -1e-3f;
  float u = ix / z, v = iy / z;
  float gx = u / (WFF * 0.5f) - 1.0f;
  float gy = v / (HFF * 0.5f) - 1.0f;
  float xpix = (gx + 1.0f) * 0.5f * (WFF - 1);
  float ypix = (gy + 1.0f) * 0.5f * (HFF - 1);
  float x0f = floorf(xpix), y0f = floorf(ypix);
  float wx1 = xpix - x0f, wx0 = 1.0f - wx1;
  float wy1 = ypix - y0f, wy0 = 1.0f - wy1;
  int x0 = (int)x0f, y0 = (int)y0f;

  int xs[2] = {x0, x0 + 1};
  int ys[2] = {y0, y0 + 1};
  float wxs[2] = {wx0, wx1};
  float wys[2] = {wy0, wy1};
  int   off[4];
  float wt[4];
#pragma unroll
  for (int yi = 0; yi < 2; ++yi)
#pragma unroll
    for (int xi = 0; xi < 2; ++xi) {
      int qd = yi * 2 + xi;
      bool valid = (xs[xi] >= 0) && (xs[xi] <= WFF - 1) &&
                   (ys[yi] >= 0) && (ys[yi] <= HFF - 1);
      int xc = xs[xi] < 0 ? 0 : (xs[xi] > WFF - 1 ? WFF - 1 : xs[xi]);
      int yc = ys[yi] < 0 ? 0 : (ys[yi] > HFF - 1 ? HFF - 1 : ys[yi]);
      off[qd] = yc * WFF + xc;
      wt[qd] = valid ? wxs[xi] * wys[yi] : 0.0f;
    }

  const float* fb = feat + ((size_t)(k * BB + b)) * (TT * CC) * (HFF * WFF);
  _Float16* ob = outf + (size_t)(b * NN + n) * FEATD + k * (TT * CC * 8) + corner;
  for (int tc = 0; tc < TT * CC; ++tc) {
    const float* fp = fb + (size_t)tc * (HFF * WFF);
    float val = wt[0] * fp[off[0]] + wt[1] * fp[off[1]] +
                wt[2] * fp[off[2]] + wt[3] * fp[off[3]];
    ob[(size_t)tc * 8] = (_Float16)val;
  }
}

// ---------------------------------------------------------------------------
extern "C" void kernel_launch(void* const* d_in, const int* in_sizes, int n_in,
                              void* d_out, int out_size, void* d_ws, size_t ws_size,
                              hipStream_t stream)
{
  (void)in_sizes; (void)n_in; (void)out_size; (void)ws_size;
  const float* features = (const float*)d_in[0];
  const float* intr     = (const float*)d_in[1];
  const float* extr     = (const float*)d_in[2];
  const float* q0       = (const float*)d_in[3];
  const float* head_w1  = (const float*)d_in[4];
  const float* head_b1  = (const float*)d_in[5];
  const float* head_w2  = (const float*)d_in[6];
  const float* head_b2  = (const float*)d_in[7];
  const float* sa_wqkv  = (const float*)d_in[8];
  const float* sa_bqkv  = (const float*)d_in[9];
  const float* sa_wo    = (const float*)d_in[10];
  const float* sa_bo    = (const float*)d_in[11];
  const float* ca_wq    = (const float*)d_in[12];
  const float* ca_bq    = (const float*)d_in[13];
  const float* ca_wkv   = (const float*)d_in[14];
  const float* ca_bkv   = (const float*)d_in[15];
  const float* ca_wo    = (const float*)d_in[16];
  const float* ca_bo    = (const float*)d_in[17];
  const float* fp_w     = (const float*)d_in[18];
  const float* fp_b     = (const float*)d_in[19];
  const float* ln1_g    = (const float*)d_in[20];
  const float* ln1_b    = (const float*)d_in[21];
  const float* ln2_g    = (const float*)d_in[22];
  const float* ln2_b    = (const float*)d_in[23];
  const float* ln3_g    = (const float*)d_in[24];
  const float* ln3_b    = (const float*)d_in[25];
  const float* mlp_w1   = (const float*)d_in[26];
  const float* mlp_b1   = (const float*)d_in[27];
  const float* mlp_w2   = (const float*)d_in[28];
  const float* mlp_b2   = (const float*)d_in[29];
  float* out = (float*)d_out;

  uintptr_t base = (uintptr_t)d_ws;
  auto alloc = [&](size_t bytes) -> uintptr_t {
    uintptr_t r = base;
    base += (bytes + 255) & ~(size_t)255;
    return r;
  };
  // activations
  float*    q32   = (float*)   alloc((size_t)ROWS * DD * 4);
  _Float16* q16   = (_Float16*)alloc((size_t)ROWS * DD * 2);
  float*    h1    = (float*)   alloc((size_t)ROWS * DD * 4);
  _Float16* h1h   = (_Float16*)alloc((size_t)ROWS * DD * 2);
  _Float16* feats = (_Float16*)alloc((size_t)ROWS * FEATD * 2);
  float*    x32   = (float*)   alloc((size_t)ROWS * DD * 4);
  _Float16* x16   = (_Float16*)alloc((size_t)ROWS * DD * 2);
  float*    qkv   = (float*)   alloc((size_t)ROWS * 3 * DD * 4);
  _Float16* o16   = (_Float16*)alloc((size_t)ROWS * DD * 2);
  float*    x2    = (float*)   alloc((size_t)ROWS * DD * 4);
  float*    x3    = (float*)   alloc((size_t)ROWS * DD * 4);
  _Float16* x3h   = (_Float16*)alloc((size_t)ROWS * DD * 2);
  float*    fbuf  = (float*)   alloc((size_t)ROWS * DD * 4);
  _Float16* fh    = (_Float16*)alloc((size_t)ROWS * DD * 2);
  float*    qc    = (float*)   alloc((size_t)ROWS * DD * 4);
  float*    kvc   = (float*)   alloc((size_t)ROWS * 2 * DD * 4);
  float*    x4    = (float*)   alloc((size_t)ROWS * DD * 4);
  float*    x5    = (float*)   alloc((size_t)ROWS * DD * 4);
  _Float16* x5h   = (_Float16*)alloc((size_t)ROWS * DD * 2);
  float*    mh    = (float*)   alloc((size_t)ROWS * 4 * DD * 4);
  _Float16* mhh   = (_Float16*)alloc((size_t)ROWS * 4 * DD * 2);
  // transposed f16 weights (Npad x K)
  _Float16* w_h1t  = (_Float16*)alloc((size_t)DD * DD * 2);
  _Float16* w_h2t  = (_Float16*)alloc((size_t)64 * DD * 2);   // P=12 padded to 64
  _Float16* w_qkvt = (_Float16*)alloc((size_t)3 * DD * DD * 2);
  _Float16* w_sot  = (_Float16*)alloc((size_t)DD * DD * 2);
  _Float16* w_cqt  = (_Float16*)alloc((size_t)DD * DD * 2);
  _Float16* w_ckvt = (_Float16*)alloc((size_t)2 * DD * DD * 2);
  _Float16* w_cot  = (_Float16*)alloc((size_t)DD * DD * 2);
  _Float16* w_fpt  = (_Float16*)alloc((size_t)DD * FEATD * 2);
  _Float16* w_m1t  = (_Float16*)alloc((size_t)4 * DD * DD * 2);
  _Float16* w_m2t  = (_Float16*)alloc((size_t)DD * 4 * DD * 2);

  auto cvtT = [&](const float* W, _Float16* Wt, int K, int Nc, int Npad) {
    int tot = Npad * K;
    td_cvtT<<<(tot + 255) / 256, 256, 0, stream>>>(W, Wt, K, Nc, Npad);
  };
  auto gemm = [&](const _Float16* A, const _Float16* Bt, const float* bias,
                  const float* resid, float* C, _Float16* C16,
                  int K, int Nc, int Npad, int relu) {
    dim3 g(Npad / 64, ROWS / 16);               // 16x64 tile per wave
    td_gemm16<<<g, 32, 0, stream>>>(A, Bt, bias, resid, C, C16, ROWS, K, Nc, relu);
  };
  auto ln = [&](const float* X, const float* g, const float* b, float* Y,
                _Float16* Y16) {
    td_ln<<<ROWS / 8, 256, 0, stream>>>(X, g, b, Y, Y16, ROWS);
  };

  const float scale = 0.125f;                 // 1/sqrt(64)
  const int attn_waves = BB * NHH * NN;       // 3200 -> 400 blocks exactly

  // init q, convert shared head weights once
  td_initq<<<(BB * NN * DD + 255) / 256, 256, 0, stream>>>(q0, q32, q16);
  cvtT(head_w1, w_h1t, DD, DD, DD);
  cvtT(head_w2, w_h2t, DD, PP, 64);

  for (int i = 0; i < LL; ++i) {
    float* traj_i = out + (size_t)i * BB * NN * PP;
    // head MLP -> trajectory i (written directly into output)
    gemm(q16, w_h1t, head_b1, nullptr, h1, h1h, DD, DD, DD, 1);
    gemm(h1h, w_h2t, head_b2, nullptr, traj_i, nullptr, DD, PP, 64, 0);
    // feature sampling from trajectory
    td_sample<<<(BB * NN * KK * 8 + 255) / 256, 256, 0, stream>>>(
        traj_i, features, intr, extr, feats);
    // x = LN1(q)
    ln(q32, ln1_g + i * DD, ln1_b + i * DD, x32, x16);
    // self-attention
    cvtT(sa_wqkv + (size_t)i * DD * 3 * DD, w_qkvt, DD, 3 * DD, 3 * DD);
    gemm(x16, w_qkvt, sa_bqkv + i * 3 * DD, nullptr, qkv, nullptr, DD, 3 * DD, 3 * DD, 0);
    td_attn<<<attn_waves / 8, 256, 0, stream>>>(
        qkv, 3 * DD, 0, qkv, 3 * DD, DD, 2 * DD, o16, DD, NN, NN, scale);
    cvtT(sa_wo + (size_t)i * DD * DD, w_sot, DD, DD, DD);
    gemm(o16, w_sot, sa_bo + i * DD, x32, x2, nullptr, DD, DD, DD, 0);
    // x = LN2(x)
    ln(x2, ln2_g + i * DD, ln2_b + i * DD, x3, x3h);
    // feature projection (dominant GEMM: 400 x 8192 x 512)
    cvtT(fp_w + (size_t)i * FEATD * DD, w_fpt, FEATD, DD, DD);
    gemm(feats, w_fpt, fp_b + i * DD, nullptr, fbuf, fh, FEATD, DD, DD, 0);
    // cross-attention
    cvtT(ca_wq + (size_t)i * DD * DD, w_cqt, DD, DD, DD);
    gemm(x3h, w_cqt, ca_bq + i * DD, nullptr, qc, nullptr, DD, DD, DD, 0);
    cvtT(ca_wkv + (size_t)i * DD * 2 * DD, w_ckvt, DD, 2 * DD, 2 * DD);
    gemm(fh, w_ckvt, ca_bkv + i * 2 * DD, nullptr, kvc, nullptr, DD, 2 * DD, 2 * DD, 0);
    td_attn<<<attn_waves / 8, 256, 0, stream>>>(
        qc, DD, 0, kvc, 2 * DD, 0, DD, o16, DD, NN, NN, scale);
    cvtT(ca_wo + (size_t)i * DD * DD, w_cot, DD, DD, DD);
    gemm(o16, w_cot, ca_bo + i * DD, x3, x4, nullptr, DD, DD, DD, 0);
    // x = LN3(x)
    ln(x4, ln3_g + i * DD, ln3_b + i * DD, x5, x5h);
    // MLP + residual -> new q
    cvtT(mlp_w1 + (size_t)i * DD * 4 * DD, w_m1t, DD, 4 * DD, 4 * DD);
    gemm(x5h, w_m1t, mlp_b1 + i * 4 * DD, nullptr, mh, mhh, DD, 4 * DD, 4 * DD, 1);
    cvtT(mlp_w2 + (size_t)i * 4 * DD * DD, w_m2t, 4 * DD, DD, DD);
    gemm(mhh, w_m2t, mlp_b2 + i * DD, x5, q32, q16, 4 * DD, DD, DD, 0);
  }
  // final head -> trajectory L
  float* traj_L = out + (size_t)LL * BB * NN * PP;
  gemm(q16, w_h1t, head_b1, nullptr, h1, h1h, DD, DD, DD, 1);
  gemm(h1h, w_h2t, head_b2, nullptr, traj_L, nullptr, DD, PP, 64, 0);
}
